// MILossGaussian_60327110639626
// MI455X (gfx1250) — matrix operand order repaired and checked by reading, not compiled, not run on hardware
//
#include <hip/hip_runtime.h>
#include <hip/hip_bf16.h>
#include <float.h>
#include <math.h>

typedef __attribute__((ext_vector_type(16))) _Float16 v16h;
typedef __attribute__((ext_vector_type(8)))  float    v8f;

#define P_ELEMS 884736           // 96*96*96 (channel dim is 1)
#define NBINS   64
#define SUPER   256              // K elements staged per pipeline stage
#define NITER   8                // stages per block
#define KPB     (SUPER * NITER)  // 2048 K elements per block
#define BLOCKS_PER_SAMPLE (P_ELEMS / KPB)  // 432

union Frag16 { v16h v; unsigned u[8]; };

// ---------------------------------------------------------------------------
// CDNA5 async memory->LDS copy (ASYNCcnt-tracked), per CDNA5 ISA+08.
// vdst operand = LDS byte offset (low 32 bits of the generic shared pointer),
// vaddr = 64-bit global address.
// ---------------------------------------------------------------------------
__device__ __forceinline__ void async_load_f32_to_lds(void* lds_ptr,
                                                      const float* gaddr) {
  const unsigned lds_off = (unsigned)(unsigned long long)lds_ptr;
  asm volatile("global_load_async_to_lds_b32 %0, %1, off"
               :: "v"(lds_off), "v"(gaddr)
               : "memory");
}
__device__ __forceinline__ void wait_async0() {
  asm volatile("s_wait_asynccnt 0x0" ::: "memory");
}

// ---------------------------------------------------------------------------
// Kernel 0: (re)initialize workspace every call (ws is poisoned by harness).
// ws layout (floats): [0..7] = {tmin,tmax,smin,smax} x 2 samples (as bits),
//                     [8..8+8192) = joint hist, 2 x 64 x 64 f32.
// ---------------------------------------------------------------------------
__global__ void mi_init_kernel(unsigned* __restrict__ mm, float* __restrict__ hist) {
  const int idx = blockIdx.x * blockDim.x + threadIdx.x;
  if (idx < 8) {
    const bool isMin = ((idx & 1) == 0);
    mm[idx] = isMin ? 0x7F7FFFFFu /* +FLT_MAX */ : 0u /* 0.0f (data >= 0) */;
  }
  if (idx < 2 * NBINS * NBINS) hist[idx] = 0.0f;
}

// ---------------------------------------------------------------------------
// Kernel 1: per-sample min/max of tar and src.
// Values are in [0,1) => non-negative => float bits are monotone as unsigned.
// ---------------------------------------------------------------------------
__global__ void __launch_bounds__(256)
mi_minmax_kernel(const float* __restrict__ tar, const float* __restrict__ src,
                 unsigned* __restrict__ mm) {
  __shared__ float red[256];
  const int n = blockIdx.y;
  const float* t = tar + (size_t)n * P_ELEMS;
  const float* s = src + (size_t)n * P_ELEMS;

  float tmn = FLT_MAX, tmx = 0.0f, smn = FLT_MAX, smx = 0.0f;
  for (int i = blockIdx.x * blockDim.x + threadIdx.x; i < P_ELEMS;
       i += gridDim.x * blockDim.x) {
    const float a = t[i]; tmn = fminf(tmn, a); tmx = fmaxf(tmx, a);
    const float b = s[i]; smn = fminf(smn, b); smx = fmaxf(smx, b);
  }

  float vals[4] = {tmn, tmx, smn, smx};
  for (int q = 0; q < 4; ++q) {
    red[threadIdx.x] = vals[q];
    __syncthreads();
    for (int off = 128; off > 0; off >>= 1) {
      if (threadIdx.x < off) {
        const float o = red[threadIdx.x + off];
        red[threadIdx.x] = (q & 1) ? fmaxf(red[threadIdx.x], o)
                                   : fminf(red[threadIdx.x], o);
      }
      __syncthreads();
    }
    if (threadIdx.x == 0) {
      const unsigned bits = __float_as_uint(red[0]);
      if (q & 1) atomicMax(&mm[n * 4 + q], bits);
      else       atomicMin(&mm[n * 4 + q], bits);
    }
    __syncthreads();
  }
}

// ---------------------------------------------------------------------------
// Kernel 2: joint histogram via WMMA, software-pipelined:
//   stage i+1 raw values stream memory->LDS asynchronously (ASYNCcnt)
//   while stage i runs its 16 WMMAs/wave (issued after exp-windowing).
// D[b,c] += sum_k wtar[b,k] * wsrc[c,k], tiled 16x16x32 f16 -> f32.
// Weight scale sigma/sqrt(2pi) is omitted: it cancels in p = h / sum(h).
// ---------------------------------------------------------------------------
__global__ void __launch_bounds__(256)
mi_joint_hist_kernel(const float* __restrict__ tar, const float* __restrict__ src,
                     const float* __restrict__ mm, float* __restrict__ hist) {
  __shared__ __align__(16) float    traw[2][SUPER];   // double-buffered raw stage
  __shared__ __align__(16) float    sraw[2][SUPER];
  __shared__ __align__(16) _Float16 wtar[NBINS][SUPER];  // 32 KB
  __shared__ __align__(16) _Float16 wsrc[NBINS][SUPER];  // 32 KB

  const int n     = blockIdx.y;
  const int tid   = threadIdx.x;
  const int lane  = tid & 31;
  const int wave  = tid >> 5;
  const int tileR = wave >> 1;            // 0..3
  const int tileC = (wave & 1) << 1;      // 0 or 2 (owns tileC, tileC+1)
  const int m     = lane & 15;
  const int half  = lane >> 4;

  const float tmin = mm[n * 4 + 0], tmax = mm[n * 4 + 1];
  const float smin = mm[n * 4 + 2], smax = mm[n * 4 + 3];
  const float invT = 1.0f / (tmax - tmin + 1e-10f);
  const float invS = 1.0f / (smax - smin + 1e-10f);

  const float* t = tar + (size_t)n * P_ELEMS + (size_t)blockIdx.x * KPB;
  const float* s = src + (size_t)n * P_ELEMS + (size_t)blockIdx.x * KPB;

  // weight-compute assignment: each thread owns one bin row, 64 consecutive k
  const int   wbin = tid >> 2;                 // 0..63
  const int   kb0  = (tid & 3) << 6;           // 0,64,128,192
  const float bval = (float)wbin * (1.0f / 63.0f);
  const float offT = tmin * invT + bval;       // d = raw*invT - offT
  const float offS = smin * invS + bval;
  // exp(-d^2 / (2*sigma^2)) = exp2(COEF * d^2); 1/(2*sigma^2) = 8192
  const float COEF = -8192.0f * 1.44269504088896340f;

  v8f acc0 = {};
  v8f acc1 = {};

  // Prologue: kick off async staging of super-chunk 0 (one b32 per lane each).
  int buf = 0;
  async_load_f32_to_lds(&traw[0][tid], t + tid);
  async_load_f32_to_lds(&sraw[0][tid], s + tid);

  for (int it = 0; it < NITER; ++it) {
    wait_async0();
    __syncthreads();                    // raw[buf] fully resident

    // ---- exp windowing (the real bottleneck: 128 v_exp per thread) ----
#pragma unroll 16
    for (int j = 0; j < 64; ++j) {
      const float x = traw[buf][kb0 + j];
      const float d = __builtin_fmaf(x, invT, -offT);
      wtar[wbin][kb0 + j] = (_Float16)exp2f(COEF * d * d);
      const float y = sraw[buf][kb0 + j];
      const float e = __builtin_fmaf(y, invS, -offS);
      wsrc[wbin][kb0 + j] = (_Float16)exp2f(COEF * e * e);
    }

    // ---- prefetch next super-chunk under the WMMA phase ----
    if (it + 1 < NITER) {
      const int nk = (it + 1) * SUPER + tid;
      async_load_f32_to_lds(&traw[buf ^ 1][tid], t + nk);
      async_load_f32_to_lds(&sraw[buf ^ 1][tid], s + nk);
    }
    __syncthreads();                    // weights visible to all waves

    // ---- 8 chunks x 2 WMMAs per wave ----
#pragma unroll
    for (int ck = 0; ck < SUPER / 32; ++ck) {
      const int k0 = ck * 32;
      Frag16 a, b0, b1;
#pragma unroll
      for (int v = 0; v < 8; ++v) {
        // A 16x32 (f16): lane m = row M; VGPR v holds K pair at:
        const int ka = k0 + ((v >> 2) << 4) + (half << 3) + ((v & 3) << 1);
        a.u[v] = *(const unsigned*)&wtar[tileR * 16 + m][ka];
        // B 32x16 (f16): lane m = col N (bin c); VGPR v holds K pair at:
        const int kb = k0 + ((v >> 2) << 4) + (half << 3) + ((v & 3) << 1);
        b0.u[v] = *(const unsigned*)&wsrc[tileC * 16 + m][kb];
        b1.u[v] = *(const unsigned*)&wsrc[tileC * 16 + 16 + m][kb];
      }
      acc0 = __builtin_amdgcn_wmma_f32_16x16x32_f16(false, a.v, false, b0.v,
                                                    (short)0, acc0, false, false);
      acc1 = __builtin_amdgcn_wmma_f32_16x16x32_f16(false, a.v, false, b1.v,
                                                    (short)0, acc1, false, false);
    }
    __syncthreads();                    // before weights are overwritten
    buf ^= 1;
  }

  // C/D layout: VGPR r holds M = r + 8*half, N = lane%16
  float* h = hist + (size_t)n * NBINS * NBINS;
#pragma unroll
  for (int r = 0; r < 8; ++r) {
    const int row = tileR * 16 + half * 8 + r;
    atomicAdd(&h[row * NBINS + tileC * 16 + m],       acc0[r]);
    atomicAdd(&h[row * NBINS + (tileC + 1) * 16 + m], acc1[r]);
  }
}

// ---------------------------------------------------------------------------
// Kernel 3: entropies + final loss (O(64^2), single block).
// ---------------------------------------------------------------------------
__global__ void __launch_bounds__(256)
mi_finalize_kernel(const float* __restrict__ hist, float* __restrict__ out) {
  __shared__ float red[256];
  __shared__ float marg[NBINS];
  __shared__ float ratio[2];
  const int tid = threadIdx.x;

  for (int n = 0; n < 2; ++n) {
    const float* h = hist + n * NBINS * NBINS;

    // total
    float sum = 0.0f;
    for (int i = tid; i < NBINS * NBINS; i += 256) sum += h[i];
    red[tid] = sum; __syncthreads();
    for (int off = 128; off > 0; off >>= 1) {
      if (tid < off) red[tid] += red[tid + off];
      __syncthreads();
    }
    const float total = red[0];
    __syncthreads();
    const float inv = 1.0f / total;

    // joint entropy
    float ej = 0.0f;
    for (int i = tid; i < NBINS * NBINS; i += 256) {
      const float p = h[i] * inv;
      ej -= p * logf(p + 1e-10f);
    }
    red[tid] = ej; __syncthreads();
    for (int off = 128; off > 0; off >>= 1) {
      if (tid < off) red[tid] += red[tid + off];
      __syncthreads();
    }
    const float ent_joint = red[0];
    __syncthreads();

    // marginal over columns -> p_tar
    float ent_tar = 0.0f;
    if (tid < NBINS) {
      float pm = 0.0f;
      for (int c = 0; c < NBINS; ++c) pm += h[tid * NBINS + c];
      pm *= inv;
      marg[tid] = -pm * logf(pm + 1e-10f);
    }
    __syncthreads();
    if (tid == 0) { for (int b = 0; b < NBINS; ++b) ent_tar += marg[b]; }
    __syncthreads();

    // marginal over rows -> p_src
    if (tid < NBINS) {
      float pm = 0.0f;
      for (int b = 0; b < NBINS; ++b) pm += h[b * NBINS + tid];
      pm *= inv;
      marg[tid] = -pm * logf(pm + 1e-10f);
    }
    __syncthreads();
    if (tid == 0) {
      float ent_src = 0.0f;
      for (int b = 0; b < NBINS; ++b) ent_src += marg[b];
      ratio[n] = (ent_tar + ent_src) / ent_joint;
    }
    __syncthreads();
  }
  if (tid == 0) out[0] = -0.5f * (ratio[0] + ratio[1]);
}

// ---------------------------------------------------------------------------
extern "C" void kernel_launch(void* const* d_in, const int* in_sizes, int n_in,
                              void* d_out, int out_size, void* d_ws, size_t ws_size,
                              hipStream_t stream) {
  const float* tar = (const float*)d_in[0];
  const float* src = (const float*)d_in[1];
  float*    out    = (float*)d_out;
  unsigned* mm_u   = (unsigned*)d_ws;
  float*    mm_f   = (float*)d_ws;
  float*    hist   = (float*)d_ws + 8;

  mi_init_kernel<<<32, 256, 0, stream>>>(mm_u, hist);
  mi_minmax_kernel<<<dim3(64, 2), 256, 0, stream>>>(tar, src, mm_u);
  mi_joint_hist_kernel<<<dim3(BLOCKS_PER_SAMPLE, 2), 256, 0, stream>>>(tar, src, mm_f, hist);
  mi_finalize_kernel<<<1, 256, 0, stream>>>(hist, out);
}
